// HSI_RWKV_43731357008660
// MI455X (gfx1250) — compile-verified, matrix operand order checked by reading, not run on hardware
//
#include <hip/hip_runtime.h>
#include <hip/hip_bf16.h>
#include <math.h>
#include <stdint.h>

typedef __attribute__((ext_vector_type(16))) _Float16 v16h;
typedef __attribute__((ext_vector_type(8)))  float    v8f;
typedef __attribute__((ext_vector_type(4)))  unsigned int u32x4;
typedef __attribute__((ext_vector_type(8)))  int          i32x8;
typedef __attribute__((ext_vector_type(4)))  int          i32x4;

#define B_   8
#define H_   128
#define W_   128
#define T_   (H_ * W_)          // 16384
#define C_   128
#define R_   (B_ * T_)          // 131072 rows
#define SEG_ 128                // segments along T
#define L_   (T_ / SEG_)        // 128 elements per segment
#define CH_  32                 // rows per LDS-staged chunk
#define TS_  8                  // T splits for kmax partials

// ---- CDNA5 TDM availability probes (device pass) ---------------------------
#if defined(__has_builtin)
#  if __has_builtin(__builtin_amdgcn_tensor_load_to_lds)
#    define HAVE_TDM 1
#  endif
#endif
#ifndef HAVE_TDM
#  define HAVE_TDM 0
#endif
#if defined(__has_include)
#  if __has_include(<hip/amd_detail/amd_gfx1250_TDM.h>)
#    define TDM_6ARG 1
#  endif
#endif
#ifndef TDM_6ARG
#  define TDM_6ARG 0
#endif

__device__ __forceinline__ float sigmoidf_(float x) { return 1.0f / (1.0f + __expf(-x)); }

__device__ __forceinline__ void wait_tensorcnt0() {
#if defined(__has_builtin) && __has_builtin(__builtin_amdgcn_s_wait_tensorcnt)
  __builtin_amdgcn_s_wait_tensorcnt(0);
#else
  asm volatile("s_wait_tensorcnt 0x0" ::: "memory");
#endif
}

#if HAVE_TDM
// Build a 2D D# (ISA 8.3/8.4) for a tile of `rows` x 128 f32 starting at gsrc,
// row stride 128 floats, and launch TENSOR_LOAD_TO_LDS.
__device__ __forceinline__ void tdm_load_tile(const float* gsrc, float* lds_ptr, int rows) {
  unsigned lds_off = (unsigned)(uintptr_t)(__attribute__((address_space(3))) void*)lds_ptr;
  unsigned long long ga = (unsigned long long)(uintptr_t)gsrc;
  u32x4 g0;
  g0[0] = 1u;                                            // count=1, user mode
  g0[1] = lds_off;                                       // lds_addr (bytes)
  g0[2] = (unsigned)(ga & 0xffffffffu);                  // global_addr[31:0]
  g0[3] = (unsigned)((ga >> 32) & 0x01ffffffu) | (2u << 30);  // ga[56:32] | type=2
  i32x8 g1 = {};
  g1[0] = (int)(2u << 16);                               // data_size = 4B
  g1[1] = (int)(128u << 16);                             // tensor_dim0 = 128 (elements)
  g1[2] = 0;                                             // tensor_dim1 low16 (131072 -> 0)
  g1[3] = (int)(((unsigned)R_ >> 16) | (128u << 16));    // tensor_dim1 hi | tile_dim0=128
  g1[4] = rows;                                          // tile_dim1 = rows, tile_dim2=0
  g1[5] = 128;                                           // tensor_dim0_stride = 128
  i32x4 z4 = {};
#if TDM_6ARG
  i32x8 z8 = {};
  __builtin_amdgcn_tensor_load_to_lds(g0, g1, z4, z4, z8, 0);
#else
  __builtin_amdgcn_tensor_load_to_lds(g0, g1, z4, z4, 0);
#endif
}
#endif

// Stage a CH_ x 128 chunk of K and V into LDS (TDM when available).
__device__ __forceinline__ void stage_chunk(const float* __restrict__ gK,
                                            const float* __restrict__ gV,
                                            float* sk, float* sv) {
#if HAVE_TDM
  if (threadIdx.x < 32) {          // wave 0 drives the DMA
    tdm_load_tile(gK, sk, CH_);
    tdm_load_tile(gV, sv, CH_);
    wait_tensorcnt0();
  }
#else
  for (int j = threadIdx.x; j < CH_ * C_ / 4; j += blockDim.x) {
    ((float4*)sk)[j] = ((const float4*)gK)[j];
    ((float4*)sv)[j] = ((const float4*)gV)[j];
  }
#endif
  __syncthreads();
}

// ---------------------------------------------------------------------------
// Weight prep: pack W[d][c] (f32) into f16 B-fragments.
// [kt(4)][nt(8)][lane(32)][h(16)] halves; lane -> N, halves -> K pairs.
// ---------------------------------------------------------------------------
__global__ void prep_weights(const float* __restrict__ wk, const float* __restrict__ wv,
                             const float* __restrict__ wr, const float* __restrict__ wo,
                             _Float16* __restrict__ wf) {
  int tid = blockIdx.x * blockDim.x + threadIdx.x;   // 4*16384 total
  int w    = tid >> 14;
  int rem  = tid & 16383;
  int kt   = rem >> 12;
  int rem2 = rem & 4095;
  int nt   = rem2 >> 9;
  int rem3 = rem2 & 511;
  int lane = rem3 >> 4;
  int h    = rem3 & 15;
  int n    = nt * 16 + (lane & 15);
  int base = (lane >> 4) << 3;
  int k    = kt * 32 + base + ((h < 8) ? h : (16 + (h - 8)));
  const float* src = (w == 0) ? wk : (w == 1) ? wv : (w == 2) ? wr : wo;
  wf[tid] = (_Float16)src[n * C_ + k];
}

// A-fragment loader: X f32 row-major -> v16h per ISA 16-bit A layout.
__device__ __forceinline__ void load_afrags(const float* __restrict__ X, int mbase, int lane,
                                            v16h a[4]) {
  int m = mbase + (lane & 15);
  const float* rowp = X + (size_t)m * C_;
  int base = (lane >> 4) << 3;   // 0 or 8
#pragma unroll
  for (int kt = 0; kt < 4; ++kt) {
    const float4* q = (const float4*)(rowp + kt * 32 + base);
    float4 f0 = q[0], f1 = q[1], f2 = q[4], f3 = q[5];
    v16h t;
    t[0] = (_Float16)f0.x; t[1] = (_Float16)f0.y; t[2]  = (_Float16)f0.z; t[3]  = (_Float16)f0.w;
    t[4] = (_Float16)f1.x; t[5] = (_Float16)f1.y; t[6]  = (_Float16)f1.z; t[7]  = (_Float16)f1.w;
    t[8] = (_Float16)f2.x; t[9] = (_Float16)f2.y; t[10] = (_Float16)f2.z; t[11] = (_Float16)f2.w;
    t[12]= (_Float16)f3.x; t[13]= (_Float16)f3.y; t[14] = (_Float16)f3.z; t[15] = (_Float16)f3.w;
    a[kt] = t;
  }
}

// One wave: 16 rows x N=128 via 8 WMMA N-tiles, K=128 in 4 steps.
__device__ __forceinline__ void run_gemm_wave(const v16h a[4], const _Float16* __restrict__ Wf,
                                              float* __restrict__ Out, int mbase, int lane,
                                              bool sig) {
  int mrow = mbase + ((lane >> 4) << 3);
  int ncol = lane & 15;
#pragma unroll
  for (int nt = 0; nt < 8; ++nt) {
    v8f acc = {};
#pragma unroll
    for (int kt = 0; kt < 4; ++kt) {
      v16h b = *(const v16h*)(Wf + (size_t)(((kt * 8 + nt) * 32 + lane) * 16));
      acc = __builtin_amdgcn_wmma_f32_16x16x32_f16(false, a[kt], false, b,
                                                   (short)0, acc, false, false);
    }
#pragma unroll
    for (int i = 0; i < 8; ++i) {
      float val = acc[i];
      if (sig) val = sigmoidf_(val);
      Out[(size_t)(mrow + i) * C_ + nt * 16 + ncol] = val;
    }
  }
}

// Fused k/v/r projections (r -> sigmoid -> sr). Grid: R_/128 x 256 thr.
__global__ void proj_kvr(const float* __restrict__ X, const _Float16* __restrict__ wf,
                         float* __restrict__ K, float* __restrict__ V, float* __restrict__ SR) {
  int lane  = threadIdx.x & 31;
  int wave  = threadIdx.x >> 5;
  int mbase = blockIdx.x * 128 + wave * 16;
  v16h a[4];
  load_afrags(X, mbase, lane, a);
  run_gemm_wave(a, wf,             K,  mbase, lane, false);
  run_gemm_wave(a, wf + 16384,     V,  mbase, lane, false);
  run_gemm_wave(a, wf + 2 * 16384, SR, mbase, lane, true);
}

// Output GEMM: d_out = G * wo^T. Grid: R_/128 x 256 thr.
__global__ void out_gemm(const float* __restrict__ G, const _Float16* __restrict__ wf,
                         float* __restrict__ Out) {
  int lane  = threadIdx.x & 31;
  int wave  = threadIdx.x >> 5;
  int mbase = blockIdx.x * 128 + wave * 16;
  v16h a[4];
  load_afrags(G, mbase, lane, a);
  run_gemm_wave(a, wf + 3 * 16384, Out, mbase, lane, false);
}

// ---------------------------------------------------------------------------
// kmax over T, two stages.
// ---------------------------------------------------------------------------
__global__ void kmax_part(const float* __restrict__ K, float* __restrict__ KP) {
  int blk = blockIdx.x;
  int b  = blk / (4 * TS_);
  int cg = (blk / TS_) & 3;
  int ts = blk & (TS_ - 1);
  int lane = threadIdx.x & 31;
  int y    = threadIdx.x >> 5;
  int c = cg * 32 + lane;
  float m = -1e30f;
  int t0 = ts * (T_ / TS_);
  for (int t = t0 + y; t < t0 + (T_ / TS_); t += 8)
    m = fmaxf(m, K[((size_t)b * T_ + t) * C_ + c]);
  __shared__ float sm[256];
  sm[threadIdx.x] = m;
  __syncthreads();
  for (int s = 128; s >= 32; s >>= 1) {
    if (threadIdx.x < s) sm[threadIdx.x] = fmaxf(sm[threadIdx.x], sm[threadIdx.x + s]);
    __syncthreads();
  }
  if (threadIdx.x < 32) KP[((size_t)b * TS_ + ts) * C_ + c] = sm[threadIdx.x];
}

__global__ void kmax_final(const float* __restrict__ KP, float* __restrict__ KM) {
  int gid = blockIdx.x * blockDim.x + threadIdx.x;   // B_*C_ = 1024
  int b = gid >> 7, c = gid & 127;
  float m = -1e30f;
#pragma unroll
  for (int ts = 0; ts < TS_; ++ts) m = fmaxf(m, KP[((size_t)b * TS_ + ts) * C_ + c]);
  KM[gid] = m;
}

// ---------------------------------------------------------------------------
// Sobel edge gate: grid B_*H_ blocks x 128 threads (channels).
// ---------------------------------------------------------------------------
__global__ void edge_kernel(const float* __restrict__ X, float* __restrict__ E) {
  int b = blockIdx.x >> 7;
  int h = blockIdx.x & 127;
  int c = threadIdx.x;
  __shared__ float sb[128];
  const float* Xb = X + (size_t)b * T_ * C_;
  for (int w = 0; w < W_; ++w) {
    float p[3][3];
#pragma unroll
    for (int dh = -1; dh <= 1; ++dh)
#pragma unroll
      for (int dw = -1; dw <= 1; ++dw) {
        int hh = h + dh, ww = w + dw;
        p[dh + 1][dw + 1] = (hh < 0 || hh >= H_ || ww < 0 || ww >= W_)
                                ? 0.0f
                                : Xb[((size_t)hh * W_ + ww) * C_ + c];
      }
    float gx = (p[0][2] + 2.0f * p[1][2] + p[2][2]) - (p[0][0] + 2.0f * p[1][0] + p[2][0]);
    float gy = (p[2][0] + 2.0f * p[2][1] + p[2][2]) - (p[0][0] + 2.0f * p[0][1] + p[0][2]);
    sb[c] = sqrtf(gx * gx + gy * gy + 1e-8f);
    __syncthreads();
    for (int s = 64; s > 0; s >>= 1) {
      if (c < s) sb[c] += sb[c + s];
      __syncthreads();
    }
    if (c == 0) E[(size_t)b * T_ + h * W_ + w] = sigmoidf_(sb[0] * (1.0f / C_));
    __syncthreads();
  }
}

// ---------------------------------------------------------------------------
// WKV chunked scan, TDM-staged. dir=0 forward, dir=1 backward.
// ---------------------------------------------------------------------------
__global__ void scan_seg(const float* __restrict__ K, const float* __restrict__ V,
                         const float* __restrict__ KM, const float* __restrict__ SD,
                         float* __restrict__ CA, float* __restrict__ CB, int dir) {
  __shared__ float sk[CH_ * C_];
  __shared__ float sv[CH_ * C_];
  int b = blockIdx.x >> 7, seg = blockIdx.x & 127, c = threadIdx.x;
  float dec = __expf(-SD[c] * (1.0f / T_));
  float km  = KM[b * C_ + c];
  size_t row0 = (size_t)b * T_ + seg * L_;
  float a = 0.0f, bb = 0.0f;
  for (int ch = 0; ch < L_ / CH_; ++ch) {
    int chi = dir ? (L_ / CH_ - 1 - ch) : ch;
    stage_chunk(K + (row0 + chi * CH_) * C_, V + (row0 + chi * CH_) * C_, sk, sv);
    for (int i = 0; i < CH_; ++i) {
      int t = dir ? (CH_ - 1 - i) : i;
      float ek = __expf(sk[t * C_ + c] - km);
      a  = a * dec + ek * sv[t * C_ + c];
      bb = bb * dec + ek;
    }
    __syncthreads();
  }
  size_t ci = ((size_t)b * SEG_ + seg) * C_ + c;
  CA[ci] = a;
  CB[ci] = bb;
}

__global__ void carry_scan(float* __restrict__ CA, float* __restrict__ CB,
                           const float* __restrict__ SD, int dir) {
  int gid = blockIdx.x * blockDim.x + threadIdx.x;   // B_*C_ = 1024
  int b = gid >> 7, c = gid & 127;
  float decL = __expf(-SD[c] * ((float)L_ / T_));
  float a = 0.0f, bb = 0.0f;
  for (int i = 0; i < SEG_; ++i) {
    int s = dir ? (SEG_ - 1 - i) : i;
    size_t idx = ((size_t)b * SEG_ + s) * C_ + c;
    float ta = CA[idx], tb = CB[idx];
    CA[idx] = a;   // exclusive (incoming prefix per segment)
    CB[idx] = bb;
    a  = a * decL + ta;
    bb = bb * decL + tb;
  }
}

__global__ void apply_fwd(const float* __restrict__ K, const float* __restrict__ V,
                          const float* __restrict__ KM, const float* __restrict__ SD,
                          const float* __restrict__ CA, const float* __restrict__ CB,
                          float* __restrict__ NUM, float* __restrict__ DEN) {
  __shared__ float sk[CH_ * C_];
  __shared__ float sv[CH_ * C_];
  int b = blockIdx.x >> 7, seg = blockIdx.x & 127, c = threadIdx.x;
  float dec = __expf(-SD[c] * (1.0f / T_));
  float km  = KM[b * C_ + c];
  size_t ci = ((size_t)b * SEG_ + seg) * C_ + c;
  float a = CA[ci], bb = CB[ci];
  size_t row0 = (size_t)b * T_ + seg * L_;
  for (int ch = 0; ch < L_ / CH_; ++ch) {
    stage_chunk(K + (row0 + ch * CH_) * C_, V + (row0 + ch * CH_) * C_, sk, sv);
    for (int t = 0; t < CH_; ++t) {
      size_t idx = (row0 + ch * CH_ + t) * C_ + c;
      NUM[idx] = a;          // exclusive forward prefix
      DEN[idx] = bb;
      float ek = __expf(sk[t * C_ + c] - km);
      a  = a * dec + ek * sv[t * C_ + c];
      bb = bb * dec + ek;
    }
    __syncthreads();
  }
}

__global__ void apply_bwd_fin(const float* __restrict__ K, const float* __restrict__ V,
                              const float* __restrict__ KM, const float* __restrict__ SD,
                              const float* __restrict__ SF, const float* __restrict__ CA,
                              const float* __restrict__ CB, const float* __restrict__ DEN,
                              const float* __restrict__ SR, const float* __restrict__ E,
                              float* __restrict__ NUMG) {
  __shared__ float sk[CH_ * C_];
  __shared__ float sv[CH_ * C_];
  int b = blockIdx.x >> 7, seg = blockIdx.x & 127, c = threadIdx.x;
  float dec = __expf(-SD[c] * (1.0f / T_));
  float eu  = __expf(SF[c] * (1.0f / T_));
  float km  = KM[b * C_ + c];
  size_t ci = ((size_t)b * SEG_ + seg) * C_ + c;
  float a = CA[ci], bb = CB[ci];
  size_t row0 = (size_t)b * T_ + seg * L_;
  for (int ch = L_ / CH_ - 1; ch >= 0; --ch) {
    stage_chunk(K + (row0 + ch * CH_) * C_, V + (row0 + ch * CH_) * C_, sk, sv);
    for (int i = 0; i < CH_; ++i) {
      int t = CH_ - 1 - i;
      size_t idx = (row0 + ch * CH_ + t) * C_ + c;
      float ek  = __expf(sk[t * C_ + c] - km);
      float ekv = ek * sv[t * C_ + c];
      float numv = eu * ekv + NUMG[idx] + a;    // a_f (stored) + a_b (running)
      float denv = eu * ek + DEN[idx] + bb;
      float y = numv / denv;
      NUMG[idx] = (SR[idx] + E[(size_t)b * T_ + seg * L_ + ch * CH_ + t]) * y;  // gated y -> G
      a  = a * dec + ekv;
      bb = bb * dec + ek;
    }
    __syncthreads();
  }
}

// ---------------------------------------------------------------------------
extern "C" void kernel_launch(void* const* d_in, const int* in_sizes, int n_in,
                              void* d_out, int out_size, void* d_ws, size_t ws_size,
                              hipStream_t stream) {
  const float* x  = (const float*)d_in[0];
  const float* wk = (const float*)d_in[1];
  const float* wv = (const float*)d_in[2];
  const float* wr = (const float*)d_in[3];
  const float* wo = (const float*)d_in[4];
  const float* sd = (const float*)d_in[5];
  const float* sf = (const float*)d_in[6];

  const size_t N = (size_t)R_ * C_;   // 16,777,216
  float* ws    = (float*)d_ws;
  float* k     = ws;
  float* v     = k + N;
  float* sr    = v + N;
  float* num   = sr + N;
  float* den   = num + N;
  float* kmaxp = den + N;                               // B_*TS_*C_ = 8192
  float* kmax  = kmaxp + (size_t)B_ * TS_ * C_;         // 1024
  float* edge  = kmax + (size_t)B_ * C_;                // B_*T_ = 131072
  float* cA    = edge + (size_t)B_ * T_;                // B_*SEG_*C_ = 131072
  float* cB    = cA + (size_t)B_ * SEG_ * C_;
  _Float16* wf = (_Float16*)(cB + (size_t)B_ * SEG_ * C_);  // 4*16384 halves

  // 1. Pack weights to f16 fragments.
  prep_weights<<<256, 256, 0, stream>>>(wk, wv, wr, wo, wf);
  // 2. Projections (WMMA): k, v, sr=sigmoid(r).
  proj_kvr<<<R_ / 128, 256, 0, stream>>>(x, wf, k, v, sr);
  // 3. kmax over T.
  kmax_part<<<B_ * 4 * TS_, 256, 0, stream>>>(k, kmaxp);
  kmax_final<<<4, 256, 0, stream>>>(kmaxp, kmax);
  // 4. Sobel edge gate.
  edge_kernel<<<B_ * H_, 128, 0, stream>>>(x, edge);
  // 5. Forward chunked scan (TDM-staged LDS tiles).
  scan_seg<<<B_ * SEG_, C_, 0, stream>>>(k, v, kmax, sd, cA, cB, 0);
  carry_scan<<<4, 256, 0, stream>>>(cA, cB, sd, 0);
  apply_fwd<<<B_ * SEG_, C_, 0, stream>>>(k, v, kmax, sd, cA, cB, num, den);
  // 6. Backward chunked scan + finalize g = (sr+edge)*y into num.
  scan_seg<<<B_ * SEG_, C_, 0, stream>>>(k, v, kmax, sd, cA, cB, 1);
  carry_scan<<<4, 256, 0, stream>>>(cA, cB, sd, 1);
  apply_bwd_fin<<<B_ * SEG_, C_, 0, stream>>>(k, v, kmax, sd, sf, cA, cB, den, sr, edge, num);
  // 7. Output GEMM (WMMA) into d_out.
  out_gemm<<<R_ / 128, 256, 0, stream>>>(num, wf, (float*)d_out);
  (void)in_sizes; (void)n_in; (void)out_size; (void)ws_size;
}
